// ResidualSTU_24867860644445
// MI455X (gfx1250) — matrix operand order repaired and checked by reading, not compiled, not run on hardware
//
#include <hip/hip_runtime.h>

// ---------------------------------------------------------------------------
// ResidualSTU forward for MI455X (gfx1250, wave32, WMMA).
// All heavy math runs through v_wmma_f32_16x16x32_bf16 (bf16 in, f32 accum).
// Causal per-channel FFT conv == per-channel lower-triangular Toeplitz GEMM
// with N=16 (8 batch cols plus-branch + 8 sign-alternating minus-branch).
// K-loops are branch-free and software-pipelined (fragment rotation), waves
// compute 32x16 tiles (two WMMAs share one B fragment).
// ---------------------------------------------------------------------------

#define BSZ 8
#define SL  2048
#define DM  256
#define KF  24
#define NLAY 4
#define HD  1024
#define MROWS (BSZ*SL)    // 16384
#define NCH  (SL/32)      // 64 k-chunks per channel

typedef __attribute__((ext_vector_type(16))) __bf16 v16bf;
typedef __attribute__((ext_vector_type(8)))  float  v8f;
typedef __attribute__((ext_vector_type(4)))  unsigned int v4u;

union V16 { v16bf v; unsigned short u[16]; v4u q[2]; };

static __device__ __forceinline__ unsigned short f32_to_bf16(float f) {
  unsigned int u = __float_as_uint(f);
  u = u + 0x7FFFu + ((u >> 16) & 1u);     // round-to-nearest-even
  return (unsigned short)(u >> 16);
}

// ---- fragment loaders (ISA 7.12.2 16-bit layouts, wave32) ------------------
// A: 16x32 (MxK) row-major, lda elements -> two b128 loads per lane.
static __device__ __forceinline__ void load_a_frag(V16& a,
    const unsigned short* __restrict__ A, int lda, int m0, int k0, int lane) {
  const int m  = m0 + (lane & 15);
  const int kb = (lane >> 4) << 3;                 // 0 or 8
  const unsigned short* row = A + (size_t)m * lda + k0 + kb;
  a.q[0] = *(const v4u*)(row);
  a.q[1] = *(const v4u*)(row + 16);
}

// B from transposed weights BT (N rows, K cols) -> two b128 loads per lane.
static __device__ __forceinline__ void load_bT_frag(V16& b,
    const unsigned short* __restrict__ BT, int ldbt, int k0, int n0, int lane) {
  const int kb = (lane >> 4) << 3;                 // 0 or 8
  const unsigned short* row = BT + (size_t)(n0 + (lane & 15)) * ldbt + k0 + kb;
  b.q[0] = *(const v4u*)(row);
  b.q[1] = *(const v4u*)(row + 16);
}

static __device__ __forceinline__ v8f wmma_bf16(const V16& a, const V16& b, v8f c) {
  return __builtin_amdgcn_wmma_f32_16x16x32_bf16(
      false, a.v, false, b.v, (short)0, c, false, false);
}

// Pipelined 32x16 GEMM core: two M-strips share each B fragment.
static __device__ __forceinline__ void gemm_32x16(
    v8f& acc0, v8f& acc1,
    const unsigned short* __restrict__ A, int lda, int m0,
    const unsigned short* __restrict__ BT, int ldbt, int n0, int K, int lane) {
  V16 a0, a1, b0;
  load_a_frag(a0, A, lda, m0,      0, lane);
  load_a_frag(a1, A, lda, m0 + 16, 0, lane);
  load_bT_frag(b0, BT, ldbt, 0, n0, lane);
  for (int k0 = 32; k0 < K; k0 += 32) {
    V16 na0, na1, nb;
    load_a_frag(na0, A, lda, m0,      k0, lane);
    load_a_frag(na1, A, lda, m0 + 16, k0, lane);
    load_bT_frag(nb, BT, ldbt, k0, n0, lane);
    acc0 = wmma_bf16(a0, b0, acc0);
    acc1 = wmma_bf16(a1, b0, acc1);
    a0 = na0; a1 = na1; b0 = nb;
  }
  acc0 = wmma_bf16(a0, b0, acc0);
  acc1 = wmma_bf16(a1, b0, acc1);
}

// ---------------------------------------------------------------------------
__global__ void k_cast(const float* __restrict__ in, unsigned short* __restrict__ out,
                       size_t n) {
  size_t i = (size_t)blockIdx.x * blockDim.x + threadIdx.x;
  if (i < n) out[i] = f32_to_bf16(in[i]);
}

// transpose-cast: in (R,C) f32 row-major -> out (C,R) bf16 row-major
__global__ void k_castT(const float* __restrict__ in, unsigned short* __restrict__ out,
                        int R, int C) {
  int i = blockIdx.x * blockDim.x + threadIdx.x;
  if (i < R * C) {
    int r = i / C, c = i - r * C;
    out[(size_t)c * R + r] = f32_to_bf16(in[i]);
  }
}

// RMSNorm over D=256, one row per block; optional z := x copy.
__global__ void k_rms(const float* __restrict__ x, const float* __restrict__ w,
                      unsigned short* __restrict__ xn, float* __restrict__ z,
                      int copyz) {
  __shared__ float red[DM];
  const int row = blockIdx.x, tid = threadIdx.x;
  const size_t idx = (size_t)row * DM + tid;
  float v = x[idx];
  if (copyz) z[idx] = v;
  red[tid] = v * v;
  __syncthreads();
  for (int s = DM / 2; s > 0; s >>= 1) {
    if (tid < s) red[tid] += red[tid + s];
    __syncthreads();
  }
  const float scale = rsqrtf(red[0] * (1.0f / DM) + 1e-6f);
  xn[idx] = f32_to_bf16(v * scale * w[tid]);
}

// vpT[d][s] = bf16( sum_k phi[s,k] * Mf[k,d] )
__global__ void k_vp(const float* __restrict__ phi, const float* __restrict__ Mf,
                     unsigned short* __restrict__ vpT) {
  const int i = blockIdx.x * blockDim.x + threadIdx.x;   // over D*S
  const int d = i >> 11, s = i & (SL - 1);
  float acc = 0.f;
  #pragma unroll
  for (int k = 0; k < KF; ++k) acc += phi[(size_t)s * KF + k] * Mf[(size_t)k * DM + d];
  vpT[(size_t)d * SL + s] = f32_to_bf16(acc);
}

// Pack conv RHS, k-chunk tiled: U[d][t/32][n][t%32]; n<8 plus, n>=8 times sgn[t].
__global__ void k_pack(const unsigned short* __restrict__ xp,
                       unsigned short* __restrict__ U) {
  const int i = blockIdx.x * blockDim.x + threadIdx.x;   // over D*S
  const int d = i >> 11, t = i & (SL - 1);
  const size_t base = (((size_t)d * NCH + (t >> 5)) * 16) * 32 + (t & 31);
  #pragma unroll
  for (int b = 0; b < BSZ; ++b) {
    unsigned short u = xp[((size_t)b * SL + t) * DM + d];
    U[base + (size_t)b * 32]       = u;
    U[base + (size_t)(8 + b) * 32] = (t & 1) ? (unsigned short)(u ^ 0x8000u) : u;
  }
}

// Causal Toeplitz conv as WMMA GEMM. Block = 4 waves; each wave owns two
// adjacent 16-row s-strips of one channel d and shares each B fragment.
// Filter staged reversed + zero-padded in LDS -> branch-free causal taps
// (taps with t > s land in the zero pad and contribute nothing).
__global__ void k_conv(const unsigned short* __restrict__ vpT,
                       const unsigned short* __restrict__ U,
                       float* __restrict__ x) {
  __shared__ unsigned short rp[SL + 96];     // rp[i] = vp[S-1-i]; pad = 0
  const int lane = threadIdx.x;
  const int d = blockIdx.y;
  {
    const int tid = threadIdx.y * 32 + lane; // 128 threads
    const unsigned short* vrow = vpT + (size_t)d * SL;
    for (int i = tid; i < SL; i += 128) rp[(SL - 1) - i] = vrow[i];
    if (tid < 96) rp[SL + tid] = 0;
  }
  __syncthreads();

  const int wpair = blockIdx.x * blockDim.y + threadIdx.y;  // 2-strip unit
  const int s0 = wpair * 32;                 // strips [s0,s0+16) and [s0+16,s0+32)
  const int m = lane & 15;
  const int roffLo = (SL - 1) - (s0 + m);        // strip 0: vp[s-t] == rp[roff+t]
  const int roffHi = roffLo - 16;                // strip 1
  const int kbA = (lane >> 4) << 3;
  const int kbB = (lane >> 4) << 3;
  const unsigned short* Ud = U + ((size_t)blockIdx.y * NCH * 16 + m) * 32 + kbB;

  v8f acc0 = {}, acc1 = {};
  const int chunks = (s0 + 32 + 31) >> 5;    // covers upper strip; pad guards lower

  V16 a0, a1, b0;
  #pragma unroll
  for (int i = 0; i < 8; ++i) {
    a0.u[i] = rp[roffLo + kbA + i];      a0.u[8 + i] = rp[roffLo + kbA + 16 + i];
    a1.u[i] = rp[roffHi + kbA + i];      a1.u[8 + i] = rp[roffHi + kbA + 16 + i];
  }
  b0.q[0] = *(const v4u*)(Ud);
  b0.q[1] = *(const v4u*)(Ud + 16);

  for (int c0 = 1; c0 < chunks; ++c0) {
    const int t0 = c0 << 5;
    V16 na0, na1, nb;
    #pragma unroll
    for (int i = 0; i < 8; ++i) {
      na0.u[i] = rp[roffLo + t0 + kbA + i];  na0.u[8 + i] = rp[roffLo + t0 + kbA + 16 + i];
      na1.u[i] = rp[roffHi + t0 + kbA + i];  na1.u[8 + i] = rp[roffHi + t0 + kbA + 16 + i];
    }
    const unsigned short* rb = Ud + (size_t)c0 * 16 * 32;
    nb.q[0] = *(const v4u*)(rb);
    nb.q[1] = *(const v4u*)(rb + 16);
    acc0 = wmma_bf16(a0, b0, acc0);
    acc1 = wmma_bf16(a1, b0, acc1);
    a0 = na0; a1 = na1; b0 = nb;
  }
  acc0 = wmma_bf16(a0, b0, acc0);
  acc1 = wmma_bf16(a1, b0, acc1);

  // recombine: lane c holds plus(b=c), lane c+8 holds minus(b=c); swap via xor-8
  const int nc = lane & 15;
  const int mo = (lane >> 4) * 8;
  #pragma unroll
  for (int strip = 0; strip < 2; ++strip) {
    const int sbase = s0 + strip * 16;
    #pragma unroll
    for (int r = 0; r < 8; ++r) {
      float val   = strip ? acc1[r] : acc0[r];
      float other = __shfl_xor(val, 8, 32);
      if ((lane & 8) == 0) {
        const int b = nc & 7;
        const int s = sbase + mo + r;
        const float sg = (s & 1) ? -1.0f : 1.0f;
        const size_t idx = ((size_t)b * SL + s) * DM + d;   // tile-exclusive
        x[idx] += val + sg * other;
      }
    }
  }
}

// ---- GEMMs: A(M,K) bf16 row-major @ BT(N,K) bf16 (pre-transposed) ----------
__global__ void k_gemm_f32(const unsigned short* __restrict__ A,
                           const unsigned short* __restrict__ BT,
                           float* __restrict__ C, int K, int N) {
  const int lane = threadIdx.x;
  const int m0 = blockIdx.x * 32;
  const int n0 = (blockIdx.y * blockDim.y + threadIdx.y) * 16;
  v8f acc0 = {}, acc1 = {};
  gemm_32x16(acc0, acc1, A, K, m0, BT, K, n0, K, lane);
  const int c = lane & 15, mo = (lane >> 4) * 8;
  #pragma unroll
  for (int r = 0; r < 8; ++r) {
    C[(size_t)(m0 + mo + r) * N + n0 + c]      = acc0[r];
    C[(size_t)(m0 + 16 + mo + r) * N + n0 + c] = acc1[r];
  }
}

__global__ void k_gemm_bf16(const unsigned short* __restrict__ A,
                            const unsigned short* __restrict__ BT,
                            unsigned short* __restrict__ C, int K, int N) {
  const int lane = threadIdx.x;
  const int m0 = blockIdx.x * 32;
  const int n0 = (blockIdx.y * blockDim.y + threadIdx.y) * 16;
  v8f acc0 = {}, acc1 = {};
  gemm_32x16(acc0, acc1, A, K, m0, BT, K, n0, K, lane);
  const int c = lane & 15, mo = (lane >> 4) * 8;
  #pragma unroll
  for (int r = 0; r < 8; ++r) {
    C[(size_t)(m0 + mo + r) * N + n0 + c]      = f32_to_bf16(acc0[r]);
    C[(size_t)(m0 + 16 + mo + r) * N + n0 + c] = f32_to_bf16(acc1[r]);
  }
}

// fc1 + silu gate fused: g = y * silu(gate). W1T is (2H, K): y rows n0..,
// gate rows H+n0..; 32x(16|16) tile: 2 A frags x 2 B frags -> 4 WMMA/chunk.
__global__ void k_gemm_gate(const unsigned short* __restrict__ A,
                            const unsigned short* __restrict__ W1T,
                            unsigned short* __restrict__ g, int K, int H) {
  const int lane = threadIdx.x;
  const int m0 = blockIdx.x * 32;
  const int n0 = (blockIdx.y * blockDim.y + threadIdx.y) * 16;
  v8f accy0 = {}, accy1 = {}, accg0 = {}, accg1 = {};
  V16 a0, a1, by, bg;
  load_a_frag(a0, A, K, m0,      0, lane);
  load_a_frag(a1, A, K, m0 + 16, 0, lane);
  load_bT_frag(by, W1T, K, 0, n0,     lane);
  load_bT_frag(bg, W1T, K, 0, H + n0, lane);
  for (int k0 = 32; k0 < K; k0 += 32) {
    V16 na0, na1, nby, nbg;
    load_a_frag(na0, A, K, m0,      k0, lane);
    load_a_frag(na1, A, K, m0 + 16, k0, lane);
    load_bT_frag(nby, W1T, K, k0, n0,     lane);
    load_bT_frag(nbg, W1T, K, k0, H + n0, lane);
    accy0 = wmma_bf16(a0, by, accy0);
    accy1 = wmma_bf16(a1, by, accy1);
    accg0 = wmma_bf16(a0, bg, accg0);
    accg1 = wmma_bf16(a1, bg, accg1);
    a0 = na0; a1 = na1; by = nby; bg = nbg;
  }
  accy0 = wmma_bf16(a0, by, accy0);
  accy1 = wmma_bf16(a1, by, accy1);
  accg0 = wmma_bf16(a0, bg, accg0);
  accg1 = wmma_bf16(a1, bg, accg1);

  const int c = lane & 15, mo = (lane >> 4) * 8;
  #pragma unroll
  for (int r = 0; r < 8; ++r) {
    float yv = accy0[r], gv = accg0[r];
    g[(size_t)(m0 + mo + r) * H + n0 + c] =
        f32_to_bf16(yv * gv / (1.0f + __expf(-gv)));
    yv = accy1[r]; gv = accg1[r];
    g[(size_t)(m0 + 16 + mo + r) * H + n0 + c] =
        f32_to_bf16(yv * gv / (1.0f + __expf(-gv)));
  }
}

// x += G @ W2 (+ z for the doubled residual); W2T is (N,K)
__global__ void k_gemm_addxz(const unsigned short* __restrict__ G,
                             const unsigned short* __restrict__ W2T,
                             float* __restrict__ x, const float* __restrict__ z,
                             int K, int N, int addz) {
  const int lane = threadIdx.x;
  const int m0 = blockIdx.x * 32;
  const int n0 = (blockIdx.y * blockDim.y + threadIdx.y) * 16;
  v8f acc0 = {}, acc1 = {};
  gemm_32x16(acc0, acc1, G, K, m0, W2T, K, n0, K, lane);
  const int c = lane & 15, mo = (lane >> 4) * 8;
  #pragma unroll
  for (int r = 0; r < 8; ++r) {
    size_t i0 = (size_t)(m0 + mo + r) * N + n0 + c;
    size_t i1 = (size_t)(m0 + 16 + mo + r) * N + n0 + c;
    float v0 = x[i0] + acc0[r];
    float v1 = x[i1] + acc1[r];
    if (addz) { v0 += z[i0]; v1 += z[i1]; }
    x[i0] = v0;
    x[i1] = v1;
  }
}

// ---------------------------------------------------------------------------
extern "C" void kernel_launch(void* const* d_in, const int* in_sizes, int n_in,
                              void* d_out, int out_size, void* d_ws, size_t ws_size,
                              hipStream_t stream) {
  const float* inputs  = (const float*)d_in[0];
  const float* W_in    = (const float*)d_in[1];
  const float* W_out   = (const float*)d_in[2];
  const float* rms1_w  = (const float*)d_in[3];
  const float* rms2_w  = (const float*)d_in[4];
  const float* M_in    = (const float*)d_in[5];
  const float* M_filt  = (const float*)d_in[6];
  const float* fc1_W   = (const float*)d_in[7];
  const float* fc2_W   = (const float*)d_in[8];
  const float* phi     = (const float*)d_in[9];

  char* w = (char*)d_ws;
  auto alloc = [&](size_t bytes) {
    char* p = w;
    w += (bytes + 255) & ~(size_t)255;
    return p;
  };
  float*          x    = (float*)alloc((size_t)MROWS * DM * 4);
  float*          z    = (float*)alloc((size_t)MROWS * DM * 4);
  unsigned short* xn   = (unsigned short*)alloc((size_t)MROWS * DM * 2);
  unsigned short* xp   = (unsigned short*)alloc((size_t)MROWS * DM * 2);
  unsigned short* U    = (unsigned short*)alloc((size_t)DM * 16 * SL * 2);
  unsigned short* vpT  = (unsigned short*)alloc((size_t)DM * SL * 2);
  unsigned short* g    = (unsigned short*)alloc((size_t)MROWS * HD * 2);
  unsigned short* inb  = (unsigned short*)alloc((size_t)MROWS * DM * 2);
  unsigned short* WinT = (unsigned short*)alloc((size_t)DM * DM * 2);
  unsigned short* WoutT= (unsigned short*)alloc((size_t)DM * DM * 2);
  unsigned short* MiT  = (unsigned short*)alloc((size_t)NLAY * DM * DM * 2);
  unsigned short* fc1T = (unsigned short*)alloc((size_t)NLAY * 2 * HD * DM * 2);
  unsigned short* fc2T = (unsigned short*)alloc((size_t)NLAY * DM * HD * 2);

  auto cast = [&](const float* src, unsigned short* dst, size_t n) {
    k_cast<<<(unsigned)((n + 255) / 256), 256, 0, stream>>>(src, dst, n);
  };
  auto castT = [&](const float* src, unsigned short* dst, int R, int C) {
    k_castT<<<(unsigned)(((size_t)R * C + 255) / 256), 256, 0, stream>>>(src, dst, R, C);
  };
  cast(inputs, inb, (size_t)MROWS * DM);
  castT(W_in,  WinT,  DM, DM);
  castT(W_out, WoutT, DM, DM);
  for (int l = 0; l < NLAY; ++l) {
    castT(M_in  + (size_t)l * DM * DM,     MiT  + (size_t)l * DM * DM,     DM, DM);
    castT(fc1_W + (size_t)l * DM * 2 * HD, fc1T + (size_t)l * 2 * HD * DM, DM, 2 * HD);
    castT(fc2_W + (size_t)l * HD * DM,     fc2T + (size_t)l * DM * HD,     HD, DM);
  }

  const dim3 gb(32, 4);
  // x = inputs @ W_in
  k_gemm_f32<<<dim3(MROWS / 32, (DM / 16) / 4), gb, 0, stream>>>(inb, WinT, x, DM, DM);

  for (int l = 0; l < NLAY; ++l) {
    // --- STU branch ---
    k_rms<<<MROWS, DM, 0, stream>>>(x, rms1_w + (size_t)l * DM, xn, z, 1);
    k_gemm_bf16<<<dim3(MROWS / 32, (DM / 16) / 4), gb, 0, stream>>>(
        xn, MiT + (size_t)l * DM * DM, xp, DM, DM);
    k_vp<<<(DM * SL) / 256, 256, 0, stream>>>(phi, M_filt + (size_t)l * KF * DM, vpT);
    k_pack<<<(DM * SL) / 256, 256, 0, stream>>>(xp, U);
    k_conv<<<dim3((SL / 32) / 4, DM), gb, 0, stream>>>(vpT, U, x);
    // --- MLP branch ---
    k_rms<<<MROWS, DM, 0, stream>>>(x, rms2_w + (size_t)l * DM, xn, z, 0);
    k_gemm_gate<<<dim3(MROWS / 32, (HD / 16) / 4), gb, 0, stream>>>(
        xn, fc1T + (size_t)l * 2 * HD * DM, g, DM, HD);
    k_gemm_addxz<<<dim3(MROWS / 32, (DM / 16) / 4), gb, 0, stream>>>(
        g, fc2T + (size_t)l * DM * HD, x, z, HD, DM, 1);
  }

  // out = x @ W_out
  cast(x, xn, (size_t)MROWS * DM);  // reuse xn as bf16(x)
  k_gemm_f32<<<dim3(MROWS / 32, (DM / 16) / 4), gb, 0, stream>>>(
      xn, WoutT, (float*)d_out, DM, DM);
}
